// SelfAttention_12163347382360
// MI455X (gfx1250) — matrix operand order repaired and checked by reading, not compiled, not run on hardware
//
#include <hip/hip_runtime.h>
#include <cstdint>

#define B_ 8
#define S_ 1024
#define D_ 1024
#define H_ 16
#define DH_ 64
#define MROWS (B_ * S_) /* 8192 */
#define MASK_VAL (-2147483648.0f)

typedef __bf16 v16bf __attribute__((ext_vector_type(16)));
typedef __bf16 v8bf  __attribute__((ext_vector_type(8)));
typedef float  v8f   __attribute__((ext_vector_type(8)));

static __device__ __forceinline__ unsigned short bf16_bits(float f) {
  unsigned u = __builtin_bit_cast(unsigned, f);
  u += 0x7fffu + ((u >> 16) & 1u); // round-to-nearest-even
  return (unsigned short)(u >> 16);
}
static __device__ __forceinline__ __bf16 f2bf(float f) {
  unsigned short s = bf16_bits(f);
  return __builtin_bit_cast(__bf16, s);
}

static __device__ __forceinline__ v8f wmma_bf16(v16bf a, v16bf b, v8f c) {
  // D = A(16x32) * B(32x16) + C, f32 accum
  return __builtin_amdgcn_wmma_f32_16x16x32_bf16(false, a, false, b, (short)0, c,
                                                 false, false);
}

// A fragment (16x32 bf16): caller passes p = base + row*ld + (lane>=16 ? 8 : 0)
static __device__ __forceinline__ v16bf frag_a(const __bf16* p) {
  union { v16bf v; v8bf h[2]; } u;
  u.h[0] = *(const v8bf*)(p);
  u.h[1] = *(const v8bf*)(p + 16);
  return u.v;
}
// B fragment (32x16 bf16): caller passes p = base + col*ld + (lane>=16 ? 8 : 0)
static __device__ __forceinline__ v16bf frag_b(const __bf16* p) {
  union { v16bf v; v8bf h[2]; } u;
  u.h[0] = *(const v8bf*)(p);
  u.h[1] = *(const v8bf*)(p + 16);
  return u.v;
}

static __device__ __forceinline__ v8f vzero8() {
  v8f z = {0.f, 0.f, 0.f, 0.f, 0.f, 0.f, 0.f, 0.f};
  return z;
}

// ---------------------------------------------------------------------------
// 16-lane (DPP16 row) reductions via row_ror rotations: pure VALU, no LDS.
// Rows of 16 lanes exactly match the WMMA C-layout half-wave row domains.
// ---------------------------------------------------------------------------
template <int CTRL>
static __device__ __forceinline__ float dpp_f32(float x) {
  int r = __builtin_amdgcn_update_dpp(0, __builtin_bit_cast(int, x), CTRL,
                                      0xf, 0xf, true);
  return __builtin_bit_cast(float, r);
}
static __device__ __forceinline__ float rowmax16(float x) {
  x = fmaxf(x, dpp_f32<0x121>(x)); // row_ror:1
  x = fmaxf(x, dpp_f32<0x122>(x)); // row_ror:2
  x = fmaxf(x, dpp_f32<0x124>(x)); // row_ror:4
  x = fmaxf(x, dpp_f32<0x128>(x)); // row_ror:8
  return x;
}
static __device__ __forceinline__ float rowsum16(float x) {
  x += dpp_f32<0x121>(x);
  x += dpp_f32<0x122>(x);
  x += dpp_f32<0x124>(x);
  x += dpp_f32<0x128>(x);
  return x;
}

// ---------------------------------------------------------------------------
// X (f32) -> bf16, straight copy-cast, float4 in / 8B out
// ---------------------------------------------------------------------------
__global__ __launch_bounds__(256) void cast_f32_bf16(const float* __restrict__ x,
                                                     __bf16* __restrict__ y, int n4) {
  int i = blockIdx.x * 256 + threadIdx.x;
  if (i >= n4) return;
  float4 v = ((const float4*)x)[i];
  unsigned lo = (unsigned)bf16_bits(v.x) | ((unsigned)bf16_bits(v.y) << 16);
  unsigned hi = (unsigned)bf16_bits(v.z) | ((unsigned)bf16_bits(v.w) << 16);
  ((uint2*)y)[i] = make_uint2(lo, hi);
}

// ---------------------------------------------------------------------------
// W [K][N] f32 -> WT [N][K] bf16 (tiled via LDS, both sides coalesced)
// ---------------------------------------------------------------------------
__global__ __launch_bounds__(256) void transpose_cast(const float* __restrict__ W,
                                                      __bf16* __restrict__ WT) {
  __shared__ float t[32][33];
  int tx = threadIdx.x, ty = threadIdx.y;
  int n = blockIdx.x * 32 + tx;
  int k0 = blockIdx.y * 32;
#pragma unroll
  for (int i = 0; i < 4; ++i)
    t[ty + i * 8][tx] = W[(size_t)(k0 + ty + i * 8) * D_ + n];
  __syncthreads();
  int ko = k0 + tx;
  int nb = blockIdx.x * 32;
#pragma unroll
  for (int i = 0; i < 4; ++i)
    WT[(size_t)(nb + ty + i * 8) * D_ + ko] = f2bf(t[tx][ty + i * 8]);
}

// ---------------------------------------------------------------------------
// C[M,N] = (A[M,K] @ W[K,N] + bias) * oscale, A bf16 row-major, W as WT[N][K].
// Block tile 128x128, 8 waves of 32x64, K-step 32 through LDS.
// MODE 0: out bf16 row-major     (Q pre-scaled by 1/sqrt(DH), K)
// MODE 1: out bf16 VT[b][n][s]   (V, transposed per batch for flash P@V)
// MODE 2: out f32 row-major      (hidden, pre-LayerNorm)
// ---------------------------------------------------------------------------
template <int MODE>
__global__ __launch_bounds__(256) void gemm_wmma(const __bf16* __restrict__ A,
                                                 const __bf16* __restrict__ BT,
                                                 const float* __restrict__ bias,
                                                 void* __restrict__ outp,
                                                 float oscale) {
  constexpr int Kd = D_;
  constexpr int Nd = D_;
  __shared__ __align__(16) __bf16 sA[128 * 40];
  __shared__ __align__(16) __bf16 sB[128 * 40];
  int tid = threadIdx.x, lane = tid & 31, wave = tid >> 5;
  int l15 = lane & 15, hi = (lane >> 4) & 1;
  int m0 = blockIdx.y * 128, n0 = blockIdx.x * 128;
  int wm = (wave & 3) * 32, wn = (wave >> 2) * 64;

  v8f acc[2][4];
#pragma unroll
  for (int i = 0; i < 2; ++i)
#pragma unroll
    for (int j = 0; j < 4; ++j) acc[i][j] = vzero8();

  for (int kk = 0; kk < Kd; kk += 32) {
    __syncthreads();
#pragma unroll
    for (int c = 0; c < 2; ++c) { // 512 x 16B chunks, 2 per thread
      int idx = tid + c * 256;
      int r = idx >> 2, q = idx & 3;
      *(int4*)&sA[r * 40 + q * 8] =
          *(const int4*)&A[(size_t)(m0 + r) * Kd + kk + q * 8];
      *(int4*)&sB[r * 40 + q * 8] =
          *(const int4*)&BT[(size_t)(n0 + r) * Kd + kk + q * 8];
    }
    __syncthreads();
    v16bf a0 = frag_a(&sA[(wm + l15) * 40 + hi * 8]);
    v16bf a1 = frag_a(&sA[(wm + 16 + l15) * 40 + hi * 8]);
#pragma unroll
    for (int j = 0; j < 4; ++j) {
      v16bf bb = frag_b(&sB[(wn + 16 * j + l15) * 40 + hi * 8]);
      acc[0][j] = wmma_bf16(a0, bb, acc[0][j]);
      acc[1][j] = wmma_bf16(a1, bb, acc[1][j]);
    }
  }
  __syncthreads(); // sA reused as per-wave staging below

  __bf16* stg = &sA[wave * 256]; // 16x16 bf16 wave-private tile
#pragma unroll
  for (int i = 0; i < 2; ++i) {
#pragma unroll
    for (int j = 0; j < 4; ++j) {
      int gmb = m0 + wm + 16 * i + 8 * hi; // first of this lane's 8 rows
      int gn = n0 + wn + 16 * j + l15;
      float bv = bias[gn];
      if constexpr (MODE == 0) {
        __bf16* out = (__bf16*)outp;
#pragma unroll
        for (int r = 0; r < 8; ++r)
          stg[(8 * hi + r) * 16 + l15] = f2bf((acc[i][j][r] + bv) * oscale);
        int srow = lane >> 1, sh = (lane & 1) * 8; // 32 lanes cover 16x16
        int4 vv = *(const int4*)&stg[srow * 16 + sh];
        *(int4*)&out[(size_t)(m0 + wm + 16 * i + srow) * Nd + n0 + wn + 16 * j + sh] = vv;
      } else if constexpr (MODE == 1) {
        __bf16* out = (__bf16*)outp; // VT [B][D][S]
        int bb = gmb / S_, ss = gmb % S_;
        unsigned pk[4];
#pragma unroll
        for (int r = 0; r < 4; ++r)
          pk[r] = (unsigned)bf16_bits((acc[i][j][2 * r] + bv) * oscale) |
                  ((unsigned)bf16_bits((acc[i][j][2 * r + 1] + bv) * oscale) << 16);
        *(int4*)&out[((size_t)bb * D_ + gn) * S_ + ss] = *(const int4*)pk;
      } else {
        float* out = (float*)outp;
#pragma unroll
        for (int r = 0; r < 8; ++r)
          out[(size_t)(gmb + r) * Nd + gn] = (acc[i][j][r] + bv) * oscale;
      }
    }
  }
}

// ---------------------------------------------------------------------------
// Flash attention: block = (b, h, 128-query strip); each wave owns 16 queries.
// Q is pre-scaled by 1/sqrt(DH) in its projection epilogue.
// Streams 64-key chunks: scores via WMMA, online softmax (DPP16 row_ror
// reductions, no LDS), P staged through wave-private LDS (C->A layout), P@V^T.
// ---------------------------------------------------------------------------
__global__ __launch_bounds__(256) void flash_attn(const __bf16* __restrict__ Q,
                                                  const __bf16* __restrict__ Kb,
                                                  const __bf16* __restrict__ VT,
                                                  const int* __restrict__ lens,
                                                  __bf16* __restrict__ ctx) {
  __shared__ __align__(16) __bf16 sP[8][16 * 80];
  int lane = threadIdx.x & 31, wave = threadIdx.x >> 5;
  int l15 = lane & 15, hi = (lane >> 4) & 1;
  int b = blockIdx.z, h = blockIdx.y;
  int q0 = blockIdx.x * 128 + wave * 16;
  int len = lens[b];

  v16bf qa[2]; // (Q/sqrt(DH)) 16x64 as two K=32 A-frags, kept resident
#pragma unroll
  for (int c = 0; c < 2; ++c)
    qa[c] = frag_a(&Q[(size_t)(b * S_ + q0 + l15) * D_ + h * DH_ + c * 32 + hi * 8]);

  float mrun[8], lrun[8];
  v8f acc[4];
#pragma unroll
  for (int r = 0; r < 8; ++r) { mrun[r] = -3.0e38f; lrun[r] = 0.f; }
#pragma unroll
  for (int j = 0; j < 4; ++j) acc[j] = vzero8();

  __bf16* myP = &sP[wave][0];

  for (int j0 = 0; j0 < len; j0 += 64) {
    v8f st[4];
#pragma unroll
    for (int sub = 0; sub < 4; ++sub) { // 16-key score tiles
      int kb = j0 + 16 * sub;
      const __bf16* kp = &Kb[(size_t)(b * S_ + kb + l15) * D_ + h * DH_ + hi * 8];
      v16bf kf0 = frag_b(kp);       // K^T columns = K rows, d contiguous
      v16bf kf1 = frag_b(kp + 32);
      v8f s = wmma_bf16(qa[0], kf0, vzero8());
      st[sub] = wmma_bf16(qa[1], kf1, s);
    }
    if (j0 + 64 > len) { // boundary chunk only: key-padding mask (uniform branch)
#pragma unroll
      for (int sub = 0; sub < 4; ++sub) {
        bool km = (j0 + 16 * sub + l15) >= len;
#pragma unroll
        for (int r = 0; r < 8; ++r)
          st[sub][r] = km ? MASK_VAL : st[sub][r];
      }
    }
    float mc[8];
#pragma unroll
    for (int r = 0; r < 8; ++r)
      mc[r] = rowmax16(
          fmaxf(fmaxf(st[0][r], st[1][r]), fmaxf(st[2][r], st[3][r])));

    float corr[8], rs[8];
#pragma unroll
    for (int r = 0; r < 8; ++r) {
      float mn = fmaxf(mrun[r], mc[r]);
      corr[r] = __expf(mrun[r] - mn);
      mrun[r] = mn;
      rs[r] = 0.f;
    }
#pragma unroll
    for (int sub = 0; sub < 4; ++sub)
#pragma unroll
      for (int r = 0; r < 8; ++r) {
        float p = __expf(st[sub][r] - mrun[r]); // masked -> exp(-2^31) == 0
        st[sub][r] = p;
        rs[r] += p;
      }
#pragma unroll
    for (int r = 0; r < 8; ++r) lrun[r] = lrun[r] * corr[r] + rowsum16(rs[r]);

    // stage P to LDS (bf16), re-read as A-fragments (wave-private, in-order DS)
#pragma unroll
    for (int sub = 0; sub < 4; ++sub)
#pragma unroll
      for (int r = 0; r < 8; ++r)
        myP[(8 * hi + r) * 80 + 16 * sub + l15] = f2bf(st[sub][r]);

    v16bf pa0 = frag_a(&myP[l15 * 80 + hi * 8]);
    v16bf pa1 = frag_a(&myP[l15 * 80 + 32 + hi * 8]);

#pragma unroll
    for (int j = 0; j < 4; ++j) { // ctx d-tiles
#pragma unroll
      for (int r = 0; r < 8; ++r) acc[j][r] *= corr[r];
      const __bf16* vp =
          &VT[((size_t)b * D_ + h * DH_ + 16 * j + l15) * S_ + j0 + hi * 8];
      acc[j] = wmma_bf16(pa0, frag_b(vp), acc[j]);
      acc[j] = wmma_bf16(pa1, frag_b(vp + 32), acc[j]);
    }
  }

  float inv[8];
#pragma unroll
  for (int r = 0; r < 8; ++r) inv[r] = (lrun[r] > 0.f) ? 1.f / lrun[r] : 0.f;
#pragma unroll
  for (int j = 0; j < 4; ++j)
#pragma unroll
    for (int r = 0; r < 8; ++r)
      myP[(8 * hi + r) * 80 + 16 * j + l15] = f2bf(acc[j][r] * inv[r]);

  int row = lane >> 1, base = (lane & 1) * 32; // 16 rows x 64 bf16, b128 stores
#pragma unroll
  for (int c4 = 0; c4 < 4; ++c4) {
    int col = base + c4 * 8;
    int4 vv = *(const int4*)&myP[row * 80 + col];
    *(int4*)&ctx[(size_t)(b * S_ + q0 + row) * D_ + h * DH_ + col] = vv;
  }
}

// ---------------------------------------------------------------------------
// out = LayerNorm(hidden + x) * w + b, one block per row
// ---------------------------------------------------------------------------
__global__ __launch_bounds__(256) void ln_residual(const float* __restrict__ hidden,
                                                   const float* __restrict__ x,
                                                   const float* __restrict__ g,
                                                   const float* __restrict__ bb,
                                                   float* __restrict__ out) {
  __shared__ float rs[256], rq[256];
  int row = blockIdx.x, t = threadIdx.x;
  float4 hv = ((const float4*)(hidden + (size_t)row * D_))[t];
  float4 xv = ((const float4*)(x + (size_t)row * D_))[t];
  float v0 = hv.x + xv.x, v1 = hv.y + xv.y, v2 = hv.z + xv.z, v3 = hv.w + xv.w;
  rs[t] = v0 + v1 + v2 + v3;
  rq[t] = v0 * v0 + v1 * v1 + v2 * v2 + v3 * v3;
  __syncthreads();
  for (int off = 128; off > 0; off >>= 1) {
    if (t < off) { rs[t] += rs[t + off]; rq[t] += rq[t + off]; }
    __syncthreads();
  }
  float u = rs[0] * (1.f / D_);
  float var = rq[0] * (1.f / D_) - u * u;
  float rstd = rsqrtf(var + 1e-12f);
  float4 gv = ((const float4*)g)[t];
  float4 bv = ((const float4*)bb)[t];
  float4 o;
  o.x = gv.x * ((v0 - u) * rstd) + bv.x;
  o.y = gv.y * ((v1 - u) * rstd) + bv.y;
  o.z = gv.z * ((v2 - u) * rstd) + bv.z;
  o.w = gv.w * ((v3 - u) * rstd) + bv.w;
  ((float4*)(out + (size_t)row * D_))[t] = o;
}

// ---------------------------------------------------------------------------
extern "C" void kernel_launch(void* const* d_in, const int* in_sizes, int n_in,
                              void* d_out, int out_size, void* d_ws, size_t ws_size,
                              hipStream_t stream) {
  (void)in_sizes; (void)n_in; (void)out_size; (void)ws_size;
  const float* x   = (const float*)d_in[0];
  const int* lens  = (const int*)d_in[1];
  const float* Wq  = (const float*)d_in[2];
  const float* bq  = (const float*)d_in[3];
  const float* Wk  = (const float*)d_in[4];
  const float* bk  = (const float*)d_in[5];
  const float* Wv  = (const float*)d_in[6];
  const float* bv  = (const float*)d_in[7];
  const float* Wo  = (const float*)d_in[8];
  const float* bo  = (const float*)d_in[9];
  const float* lnw = (const float*)d_in[10];
  const float* lnb = (const float*)d_in[11];
  float* out = (float*)d_out;

  char* ws = (char*)d_ws;
  const size_t nX = (size_t)MROWS * D_; // 8M elements
  const size_t nW = (size_t)D_ * D_;    // 1M elements
  __bf16* Xb  = (__bf16*)ws; ws += nX * 2;
  __bf16* WqT = (__bf16*)ws; ws += nW * 2;
  __bf16* WkT = (__bf16*)ws; ws += nW * 2;
  __bf16* WvT = (__bf16*)ws; ws += nW * 2;
  __bf16* WoT = (__bf16*)ws; ws += nW * 2;
  __bf16* Qb  = (__bf16*)ws; ws += nX * 2;
  __bf16* Kbb = (__bf16*)ws; ws += nX * 2;
  __bf16* VTb = (__bf16*)ws; ws += nX * 2;
  __bf16* Cb  = (__bf16*)ws; ws += nX * 2;
  float*  Hd  = (float*)ws;  ws += nX * 4;

  cast_f32_bf16<<<(int)((nX / 4 + 255) / 256), 256, 0, stream>>>(x, Xb, (int)(nX / 4));
  dim3 tb(32, 8), tg(D_ / 32, D_ / 32);
  transpose_cast<<<tg, tb, 0, stream>>>(Wq, WqT);
  transpose_cast<<<tg, tb, 0, stream>>>(Wk, WkT);
  transpose_cast<<<tg, tb, 0, stream>>>(Wv, WvT);
  transpose_cast<<<tg, tb, 0, stream>>>(Wo, WoT);

  dim3 gg(D_ / 128, MROWS / 128); // (8, 64)
  gemm_wmma<0><<<gg, 256, 0, stream>>>(Xb, WqT, bq, (void*)Qb, 0.125f); // Q/sqrt(DH)
  gemm_wmma<0><<<gg, 256, 0, stream>>>(Xb, WkT, bk, (void*)Kbb, 1.0f);
  gemm_wmma<1><<<gg, 256, 0, stream>>>(Xb, WvT, bv, (void*)VTb, 1.0f);

  flash_attn<<<dim3(S_ / 128, H_, B_), 256, 0, stream>>>(Qb, Kbb, VTb, lens, Cb);

  gemm_wmma<2><<<gg, 256, 0, stream>>>(Cb, WoT, bo, (void*)Hd, 1.0f);

  ln_residual<<<MROWS, 256, 0, stream>>>(Hd, x, lnw, lnb, out);
}
